// DeformableConv_58746562675004
// MI455X (gfx1250) — matrix operand order repaired and checked by reading, not compiled, not run on hardware
//
#include <hip/hip_runtime.h>
#include <hip/hip_bf16.h>
#include <stdint.h>

// ---------------------------------------------------------------------------
// Deformable Conv v2 for gfx1250:
//   stage 1: fused offset-conv + modulator-conv + bilinear corner precompute
//   stage 2: im2col-gather + bf16 WMMA GEMM (M=8192, N=256, K=2304)
//            BM=64, BN=256(full Co), BK=32; LDS double-buffered; async B tiles;
//            fragment-order LDS (all frag traffic ds_load_b128);
//            register double-buffered B fragments to hide LDS latency.
// ---------------------------------------------------------------------------

#define B_      2
#define C_      256
#define H_      64
#define W_      64
#define HW_     (H_ * W_)          // 4096
#define CO_     256
#define K2_     9
#define KTOT_   (C_ * K2_)         // 2304
#define NSTEP   (K2_ * (C_ / 32))  // 72 K-steps of 32

typedef __attribute__((ext_vector_type(16))) __bf16 v16bf;
typedef __attribute__((ext_vector_type(8)))  float  v8f;

union alignas(32) FragBF { v16bf v; unsigned int u[8]; uint4 q[2]; };

__device__ __forceinline__ unsigned short f2bf(float f) {
    unsigned u = __builtin_bit_cast(unsigned, f);
    unsigned r = (u + 0x7FFFu + ((u >> 16) & 1u)) >> 16;   // RNE
    return (unsigned short)r;
}

// ---------------------------------------------------------------------------
// Prep A: reg_w [Co][C][3][3] fp32  ->  Wt bf16 [Co][K] with K = k2*256 + c
// ---------------------------------------------------------------------------
__global__ void prep_weight(const float* __restrict__ rw,
                            unsigned short* __restrict__ Wt) {
    int gid = blockIdx.x * 256 + threadIdx.x;
    if (gid >= CO_ * KTOT_) return;
    int n  = gid / KTOT_;
    int k  = gid - n * KTOT_;
    int k2 = k >> 8;            // k / 256
    int c  = k & 255;           // k % 256
    Wt[n * KTOT_ + k] = f2bf(rw[n * KTOT_ + c * K2_ + k2]);
}

// ---------------------------------------------------------------------------
// Prep B: per (b,k2,pixel): run both small convs, produce 4 bilinear corner
// linear indices and 4 weights (pre-multiplied by the 2*sigmoid modulator).
// ---------------------------------------------------------------------------
__global__ __launch_bounds__(256)
void prep_offsets(const float* __restrict__ x,
                  const float* __restrict__ residual,
                  const float* __restrict__ offw,
                  const float* __restrict__ offb,
                  const float* __restrict__ modw,
                  const float* __restrict__ modb,
                  int4*   __restrict__ idx_out,
                  float4* __restrict__ wgt_out) {
    int gid = blockIdx.x * 256 + threadIdx.x;         // B*K2*HW = 73728
    if (gid >= B_ * K2_ * HW_) return;
    int p  = gid & (HW_ - 1);
    int k2 = (gid >> 12) % K2_;
    int b  = gid / (HW_ * K2_);
    int ho = p >> 6, wo = p & 63;

    const float* rb  = residual + (size_t)b * C_ * HW_;
    const float* xb  = x        + (size_t)b * C_ * HW_;
    const float* wdy = offw + (size_t)(2 * k2)     * KTOT_;
    const float* wdx = offw + (size_t)(2 * k2 + 1) * KTOT_;
    const float* wm  = modw + (size_t)k2           * KTOT_;

    int   offs[9];
    bool  val[9];
#pragma unroll
    for (int i = 0; i < 3; ++i)
#pragma unroll
        for (int j = 0; j < 3; ++j) {
            int h = ho + i - 1, w = wo + j - 1;
            bool v = (h >= 0) & (h < H_) & (w >= 0) & (w < W_);
            val[i * 3 + j]  = v;
            offs[i * 3 + j] = (v ? h : 0) * W_ + (v ? w : 0);
        }

    float ady = 0.f, adx = 0.f, am = 0.f;
    for (int c = 0; c < C_; ++c) {
        const float* rc = rb + c * HW_;
        const float* xc = xb + c * HW_;
        int wbase = c * K2_;
#pragma unroll
        for (int t = 0; t < 9; ++t) {
            float rv = val[t] ? rc[offs[t]] : 0.f;
            float xv = val[t] ? xc[offs[t]] : 0.f;
            ady += rv * wdy[wbase + t];
            adx += rv * wdx[wbase + t];
            am  += xv * wm [wbase + t];
        }
    }
    ady += offb[2 * k2];
    adx += offb[2 * k2 + 1];
    am  += modb[k2];

    float mask = 2.f / (1.f + __expf(-am));

    float py = ady + (float)(ho - 1 + (k2 / 3));
    float px = adx + (float)(wo - 1 + (k2 % 3));
    float y0f = floorf(py), x0f = floorf(px);
    float wy1 = py - y0f,  wx1 = px - x0f;
    float wy0 = 1.f - wy1, wx0 = 1.f - wx1;
    int   y0 = (int)y0f,   x0 = (int)x0f;

    int4   id;
    float4 wg;
    {
        int y, x; bool v; int yc, xc;
        y = y0;     x = x0;
        v = (y >= 0) & (y < H_) & (x >= 0) & (x < W_);
        yc = min(max(y, 0), H_ - 1); xc = min(max(x, 0), W_ - 1);
        id.x = yc * W_ + xc;  wg.x = v ? wy0 * wx0 * mask : 0.f;
        y = y0;     x = x0 + 1;
        v = (y >= 0) & (y < H_) & (x >= 0) & (x < W_);
        yc = min(max(y, 0), H_ - 1); xc = min(max(x, 0), W_ - 1);
        id.y = yc * W_ + xc;  wg.y = v ? wy0 * wx1 * mask : 0.f;
        y = y0 + 1; x = x0;
        v = (y >= 0) & (y < H_) & (x >= 0) & (x < W_);
        yc = min(max(y, 0), H_ - 1); xc = min(max(x, 0), W_ - 1);
        id.z = yc * W_ + xc;  wg.z = v ? wy1 * wx0 * mask : 0.f;
        y = y0 + 1; x = x0 + 1;
        v = (y >= 0) & (y < H_) & (x >= 0) & (x < W_);
        yc = min(max(y, 0), H_ - 1); xc = min(max(x, 0), W_ - 1);
        id.w = yc * W_ + xc;  wg.w = v ? wy1 * wx1 * mask : 0.f;
    }
    idx_out[gid] = id;
    wgt_out[gid] = wg;
}

// ---------------------------------------------------------------------------
// GEMM with software pipelining.
// Fragment-order LDS: per 16x32 tile, lane L owns bytes [L*32, L*32+32)
// holding its 8 fragment dwords in order -> frag load = 2x ds_load_b128.
//   A tile (16x32, rows m): lane = (K&16? 16:0) + m ; dword v = (K&15)>>1
//   B tile (32x16, cols n): lane = (K&16? 16:0) + n ; dword v = (K&15)>>1
// B tiles in fragment order are K-contiguous 16B runs of Wt -> direct async b128.
// ---------------------------------------------------------------------------
#define BM 64
#define TILE_US 512          // ushorts per 16x32 fragment-order tile (1 KB)

__global__ __launch_bounds__(256)
void deform_gemm(const float* __restrict__ x,
                 const unsigned short* __restrict__ Wt,
                 const int4*   __restrict__ idx,
                 const float4* __restrict__ wgt,
                 float* __restrict__ out) {
    __shared__ unsigned short Abuf[2][4 * TILE_US];    //  4 KB per buffer
    __shared__ unsigned short Bbuf[2][16 * TILE_US];   // 16 KB per buffer

    const int t    = threadIdx.x;
    const int lane = t & 31;
    const int wid  = t >> 5;
    const int wmt  = wid & 3;     // wave's M-tile (0..3)
    const int wng  = wid >> 2;    // wave's N-tile group: tiles wng*8 .. +7

    const int m0  = blockIdx.x * BM;
    const int m_l = t & 63;
    const int cg  = t >> 6;                  // 0..3 : 8-channel group
    const int m_g = m0 + m_l;
    const int b   = m_g >> 12;
    const int pix = m_g & (HW_ - 1);
    const float* xb = x + (size_t)b * C_ * HW_;

    // A staging target (fragment order): fixed per thread
    const int a_half  = cg >> 1;             // K>=16
    const int a_group = cg & 1;              // K&8
    const unsigned a_us =
        (unsigned)((m_l >> 4) * TILE_US + ((a_half << 4) + (m_l & 15)) * 16 + a_group * 8);

    v8f acc[8] = {};

    int4   id = idx[((b * K2_) << 12) + pix];
    float4 wg = wgt[((b * K2_) << 12) + pix];

    // ---- prologue: stage step 0 into buffer 0
    {
#pragma unroll
        for (int q = 0; q < 4; ++q) {
            int chunk  = q * 256 + t;            // 0..1023 : 16B chunks of B tile
            int nt     = chunk >> 6;
            int rem    = chunk & 63;
            int lane_b = rem >> 1;
            int sub    = rem & 1;
            unsigned voff = (unsigned)((((nt << 4) + (lane_b & 15)) * KTOT_ +
                                        ((lane_b >> 4) << 4) + (sub << 3)) * 2);
            unsigned loff = (unsigned)(unsigned long long)
                (&Bbuf[0][nt * TILE_US + lane_b * 16 + sub * 8]);
            asm volatile("global_load_async_to_lds_b128 %0, %1, %2"
                         :: "v"(loff), "v"(voff), "s"(Wt) : "memory");
        }
        uint4 pk;
        unsigned pkk[4];
#pragma unroll
        for (int j = 0; j < 4; ++j) {
            const float* p0 = xb + (cg * 8 + 2 * j) * HW_;
            const float* p1 = p0 + HW_;
            float v0 = wg.x * p0[id.x] + wg.y * p0[id.y] + wg.z * p0[id.z] + wg.w * p0[id.w];
            float v1 = wg.x * p1[id.x] + wg.y * p1[id.y] + wg.z * p1[id.z] + wg.w * p1[id.w];
            pkk[j] = (unsigned)f2bf(v0) | ((unsigned)f2bf(v1) << 16);
        }
        pk.x = pkk[0]; pk.y = pkk[1]; pk.z = pkk[2]; pk.w = pkk[3];
        *(uint4*)&Abuf[0][a_us] = pk;
#if defined(__has_builtin) && __has_builtin(__builtin_amdgcn_s_wait_asynccnt)
        __builtin_amdgcn_s_wait_asynccnt(0);
#else
        asm volatile("s_wait_asynccnt 0" ::: "memory");
#endif
        __syncthreads();
    }

    // ---- main pipelined loop over 72 K-steps
#pragma unroll 1
    for (int s = 0; s < NSTEP; ++s) {
        const int cur = s & 1;
        const int nxt = cur ^ 1;
        const bool has_next = (s + 1) < NSTEP;

        float g0[8], g1[8], g2[8], g3[8];
        if (has_next) {
            const int sn  = s + 1;
            const int k2n = sn >> 3;
            const int c0n = (sn & 7) << 5;
            if ((sn & 7) == 0) {
                id = idx[(((b * K2_) + k2n) << 12) + pix];
                wg = wgt[(((b * K2_) + k2n) << 12) + pix];
            }
            // async B tiles for step s+1
#pragma unroll
            for (int q = 0; q < 4; ++q) {
                int chunk  = q * 256 + t;
                int nt     = chunk >> 6;
                int rem    = chunk & 63;
                int lane_b = rem >> 1;
                int sub    = rem & 1;
                unsigned voff = (unsigned)((((nt << 4) + (lane_b & 15)) * KTOT_ +
                                            k2n * 256 + c0n +
                                            ((lane_b >> 4) << 4) + (sub << 3)) * 2);
                unsigned loff = (unsigned)(unsigned long long)
                    (&Bbuf[nxt][nt * TILE_US + lane_b * 16 + sub * 8]);
                asm volatile("global_load_async_to_lds_b128 %0, %1, %2"
                             :: "v"(loff), "v"(voff), "s"(Wt) : "memory");
            }
            // issue gather loads for step s+1 (consumed after the WMMAs)
#pragma unroll
            for (int j = 0; j < 8; ++j) {
                const float* pc = xb + (c0n + cg * 8 + j) * HW_;
                g0[j] = pc[id.x]; g1[j] = pc[id.y];
                g2[j] = pc[id.z]; g3[j] = pc[id.w];
            }
        }

        // ---- compute step s from buf[cur]
        // A frag once; B frags register-double-buffered so each WMMA can issue
        // while the next fragment's ds_load_b128 pair is still in flight.
        {
            FragBF fa;
            {
                const unsigned short* ap = &Abuf[cur][wmt * TILE_US + lane * 16];
                fa.q[0] = *(const uint4*)(ap);
                fa.q[1] = *(const uint4*)(ap + 8);
            }
            const unsigned short* bbase = &Bbuf[cur][(wng * 8) * TILE_US + lane * 16];
            FragBF fbuf[2];
            fbuf[0].q[0] = *(const uint4*)(bbase);
            fbuf[0].q[1] = *(const uint4*)(bbase + 8);
#pragma unroll
            for (int j2 = 0; j2 < 8; ++j2) {
                if (j2 < 7) {
                    const unsigned short* bp = bbase + (j2 + 1) * TILE_US;
                    fbuf[(j2 + 1) & 1].q[0] = *(const uint4*)(bp);
                    fbuf[(j2 + 1) & 1].q[1] = *(const uint4*)(bp + 8);
                }
                acc[j2] = __builtin_amdgcn_wmma_f32_16x16x32_bf16(
                    false, fa.v, false, fbuf[j2 & 1].v, (short)0, acc[j2],
                    false, false);
            }
        }

        // ---- finish staging step s+1: pack + store A tile
        if (has_next) {
            uint4 pk;
            unsigned pkk[4];
#pragma unroll
            for (int j = 0; j < 4; ++j) {
                float v0 = wg.x * g0[2 * j]     + wg.y * g1[2 * j] +
                           wg.z * g2[2 * j]     + wg.w * g3[2 * j];
                float v1 = wg.x * g0[2 * j + 1] + wg.y * g1[2 * j + 1] +
                           wg.z * g2[2 * j + 1] + wg.w * g3[2 * j + 1];
                pkk[j] = (unsigned)f2bf(v0) | ((unsigned)f2bf(v1) << 16);
            }
            pk.x = pkk[0]; pk.y = pkk[1]; pk.z = pkk[2]; pk.w = pkk[3];
            *(uint4*)&Abuf[nxt][a_us] = pk;
#if defined(__has_builtin) && __has_builtin(__builtin_amdgcn_s_wait_asynccnt)
            __builtin_amdgcn_s_wait_asynccnt(0);
#else
            asm volatile("s_wait_asynccnt 0" ::: "memory");
#endif
        }
        __syncthreads();
    }

    // ---- epilogue: C/D layout lane=(N col), VGPR r -> M = r + (lane>>4)*8
    const int col  = lane & 15;
    const int half = lane >> 4;
#pragma unroll
    for (int j2 = 0; j2 < 8; ++j2) {
        const int n_g = (wng * 8 + j2) * 16 + col;
        float* ob = out + ((size_t)(b * CO_ + n_g)) * HW_;
        const int pbase = (m0 & (HW_ - 1)) + wmt * 16 + half * 8;
#pragma unroll
        for (int r = 0; r < 8; ++r) {
            ob[pbase + r] = acc[j2][r];
        }
    }
}

// ---------------------------------------------------------------------------
extern "C" void kernel_launch(void* const* d_in, const int* in_sizes, int n_in,
                              void* d_out, int out_size, void* d_ws, size_t ws_size,
                              hipStream_t stream) {
    const float* x        = (const float*)d_in[0];
    const float* residual = (const float*)d_in[1];
    const float* offset_w = (const float*)d_in[2];
    const float* offset_b = (const float*)d_in[3];
    const float* mod_w    = (const float*)d_in[4];
    const float* mod_b    = (const float*)d_in[5];
    const float* reg_w    = (const float*)d_in[6];
    float* out = (float*)d_out;

    char* ws = (char*)d_ws;
    unsigned short* Wt  = (unsigned short*)(ws);                    // 1,179,648 B
    int4*           idx = (int4*)  (ws + 1179648);                  // 1,179,648 B
    float4*         wgt = (float4*)(ws + 2359296);                  // 1,179,648 B

    prep_weight<<<(CO_ * KTOT_ + 255) / 256, 256, 0, stream>>>(reg_w, Wt);

    prep_offsets<<<(B_ * K2_ * HW_ + 255) / 256, 256, 0, stream>>>(
        x, residual, offset_w, offset_b, mod_w, mod_b, idx, wgt);

    deform_gemm<<<dim3((B_ * HW_) / BM), 256, 0, stream>>>(x, Wt, idx, wgt, out);
}
